// TransformerWithMoE_59141699666460
// MI455X (gfx1250) — matrix operand order repaired and checked by reading, not compiled, not run on hardware
//
#include <hip/hip_runtime.h>

typedef _Float16 f16x8 __attribute__((ext_vector_type(8)));
typedef _Float16 v16h  __attribute__((ext_vector_type(16)));
typedef float    v8f   __attribute__((ext_vector_type(8)));

#define TILE_M 128
#define TILE_N 128
#define TILE_K 32
#define LDSK   40   // padded LDS row stride in f16 units (keeps 16B alignment, breaks bank patterns)

__device__ __forceinline__ v16h cat8(f16x8 lo, f16x8 hi) {
  v16h r;
#pragma unroll
  for (int j = 0; j < 8; ++j) { r[j] = lo[j]; r[j + 8] = hi[j]; }
  return r;
}

// ---------------------------------------------------------------------------
// Generic batched GEMM, fp32 in/out, f16 WMMA with fp32 accumulation.
// C[M,N] = act( alpha * A[M,Kd] @ B + bias )
//   transB=0: B is row-major [Kd,N] (ldb = row stride), staged transposed.
//   transB=1: B given as B^T row-major [N,Kd] (ldb = row stride), staged directly.
// Batch z: off = (z/binner)*s?o + (z%binner)*s?i for A, B, C.
// Requirements (all call sites satisfy): Kd % 32 == 0, M % 128 == 0,
// and N % 128 == 0 whenever transB = 1.
// Register-prefetch software pipeline: tile t+1 global loads issue before the
// WMMA compute of tile t, so s_wait_loadcnt lands one full k-step later.
// ---------------------------------------------------------------------------
__launch_bounds__(256)
__global__ void gemm_wmma_f16(const float* __restrict__ A, const float* __restrict__ B,
                              const float* __restrict__ bias, float* __restrict__ C,
                              int M, int N, int Kd, int lda, int ldb, int ldc,
                              float alpha, int transB, int relu, int binner,
                              long long sAo, long long sAi, long long sBo, long long sBi,
                              long long sCo, long long sCi)
{
  const int z = blockIdx.z;
  A += (long long)(z / binner) * sAo + (long long)(z % binner) * sAi;
  B += (long long)(z / binner) * sBo + (long long)(z % binner) * sBi;
  C += (long long)(z / binner) * sCo + (long long)(z % binner) * sCi;

  __shared__ __align__(16) _Float16 As[TILE_M * LDSK];
  __shared__ __align__(16) _Float16 Bs[TILE_N * LDSK];

  const int tid   = threadIdx.x;
  const int wave  = tid >> 5;
  const int lane  = tid & 31;
  const int wm    = wave & 1;    // 0..1  (64-row slab)
  const int wn    = wave >> 1;   // 0..3  (32-col slab)
  const int l15   = lane & 15;
  const int lhalf = lane >> 4;
  const int m0 = blockIdx.y * TILE_M;
  const int n0 = blockIdx.x * TILE_N;

  // staging index math (fixed per thread)
  const int arow = tid >> 3;            // 0..31 within a 32-row step, x4 steps -> rows 0..127
  const int acol = (tid & 7) << 2;      // 0,4,..,28
  const int krow0 = tid >> 5;           // transB=0: k row base 0..7
  const int n4    = (tid & 31) << 2;    // transB=0: n 0..124

  v8f acc[4][2];
#pragma unroll
  for (int i = 0; i < 4; ++i)
#pragma unroll
    for (int j = 0; j < 2; ++j)
#pragma unroll
      for (int r = 0; r < 8; ++r) acc[i][j][r] = 0.f;

  float4 rA[4], rB[4];

  // ---- prefetch k-tile 0 ----
#pragma unroll
  for (int i = 0; i < 4; ++i)
    rA[i] = *(const float4*)(A + (long long)(m0 + arow + i * 32) * lda + acol);
  if (transB) {
#pragma unroll
    for (int i = 0; i < 4; ++i)
      rB[i] = *(const float4*)(B + (long long)(n0 + arow + i * 32) * ldb + acol);
  } else {
#pragma unroll
    for (int i = 0; i < 4; ++i) {
      const float* srcp = B + (long long)(krow0 + i * 8) * ldb;
      int gn = n0 + n4;
      if (gn + 3 < N) {
        rB[i] = *(const float4*)(srcp + gn);
      } else {
        rB[i].x = (gn + 0 < N) ? srcp[gn + 0] : 0.f;
        rB[i].y = (gn + 1 < N) ? srcp[gn + 1] : 0.f;
        rB[i].z = (gn + 2 < N) ? srcp[gn + 2] : 0.f;
        rB[i].w = (gn + 3 < N) ? srcp[gn + 3] : 0.f;
      }
    }
  }

  for (int kk = 0; kk < Kd; kk += TILE_K) {
    __syncthreads();   // previous compute finished reading LDS
    // ---- store prefetched tile to LDS (fp32 -> f16) ----
#pragma unroll
    for (int i = 0; i < 4; ++i) {
      _Float16* d = &As[(arow + i * 32) * LDSK + acol];
      d[0] = (_Float16)rA[i].x; d[1] = (_Float16)rA[i].y;
      d[2] = (_Float16)rA[i].z; d[3] = (_Float16)rA[i].w;
    }
    if (transB) {
#pragma unroll
      for (int i = 0; i < 4; ++i) {
        _Float16* d = &Bs[(arow + i * 32) * LDSK + acol];
        d[0] = (_Float16)rB[i].x; d[1] = (_Float16)rB[i].y;
        d[2] = (_Float16)rB[i].z; d[3] = (_Float16)rB[i].w;
      }
    } else {
#pragma unroll
      for (int i = 0; i < 4; ++i) {
        int kr = krow0 + i * 8;
        Bs[(n4 + 0) * LDSK + kr] = (_Float16)rB[i].x;
        Bs[(n4 + 1) * LDSK + kr] = (_Float16)rB[i].y;
        Bs[(n4 + 2) * LDSK + kr] = (_Float16)rB[i].z;
        Bs[(n4 + 3) * LDSK + kr] = (_Float16)rB[i].w;
      }
    }
    __syncthreads();   // tile visible to all waves

    // ---- prefetch k-tile kk+32 (loads issue before the WMMAs below) ----
    const int kn = kk + TILE_K;
    if (kn < Kd) {
#pragma unroll
      for (int i = 0; i < 4; ++i)
        rA[i] = *(const float4*)(A + (long long)(m0 + arow + i * 32) * lda + kn + acol);
      if (transB) {
#pragma unroll
        for (int i = 0; i < 4; ++i)
          rB[i] = *(const float4*)(B + (long long)(n0 + arow + i * 32) * ldb + kn + acol);
      } else {
#pragma unroll
        for (int i = 0; i < 4; ++i) {
          const float* srcp = B + (long long)(kn + krow0 + i * 8) * ldb;
          int gn = n0 + n4;
          if (gn + 3 < N) {
            rB[i] = *(const float4*)(srcp + gn);
          } else {
            rB[i].x = (gn + 0 < N) ? srcp[gn + 0] : 0.f;
            rB[i].y = (gn + 1 < N) ? srcp[gn + 1] : 0.f;
            rB[i].z = (gn + 2 < N) ? srcp[gn + 2] : 0.f;
            rB[i].w = (gn + 3 < N) ? srcp[gn + 3] : 0.f;
          }
        }
      }
    }

    // ---- B fragments: lane half k-base 0/16, contiguous 16 f16 ----
    v16h bf[2];
#pragma unroll
    for (int fn = 0; fn < 2; ++fn) {
      int nl = wn * 32 + fn * 16 + l15;
      int kb = lhalf * 16;
      f16x8 lo = *(const f16x8*)&Bs[nl * LDSK + kb];
      f16x8 hi = *(const f16x8*)&Bs[nl * LDSK + kb + 8];
      bf[fn] = cat8(lo, hi);
    }
    // ---- A fragments: lane half k-base 0/16, contiguous 16 f16 ----
#pragma unroll
    for (int fm = 0; fm < 4; ++fm) {
      int ml = wm * 64 + fm * 16 + l15;
      int kb = lhalf * 16;
      f16x8 lo = *(const f16x8*)&As[ml * LDSK + kb];
      f16x8 hi = *(const f16x8*)&As[ml * LDSK + kb + 8];
      v16h af = cat8(lo, hi);
      acc[fm][0] = __builtin_amdgcn_wmma_f32_16x16x32_f16(false, af, false, bf[0],
                                                          (short)0, acc[fm][0], false, false);
      acc[fm][1] = __builtin_amdgcn_wmma_f32_16x16x32_f16(false, af, false, bf[1],
                                                          (short)0, acc[fm][1], false, false);
    }
  }

  // ---- epilogue: C layout lanes0-15 M=r, lanes16-31 M=8+r ----
#pragma unroll
  for (int fm = 0; fm < 4; ++fm)
#pragma unroll
    for (int fn = 0; fn < 2; ++fn) {
      int nG = n0 + wn * 32 + fn * 16 + l15;
      if (nG >= N) continue;
      float bv = bias ? bias[nG] : 0.f;
#pragma unroll
      for (int r = 0; r < 8; ++r) {
        int mG = m0 + wm * 64 + fm * 16 + lhalf * 8 + r;
        float val = alpha * acc[fm][fn][r] + bv;
        if (relu) val = fmaxf(val, 0.f);
        C[(long long)mG * ldc + nG] = val;
      }
    }
}

// ---------------------------------------------------------------------------
// Embedding lookup * sqrt(D) + sinusoidal PE (computed in-kernel).
// grid = 2048 token rows, block 256 (2 dims/thread).
// ---------------------------------------------------------------------------
__global__ void embed_pe(const int* __restrict__ tok, const float* __restrict__ emb,
                         float* __restrict__ out)
{
  int n = blockIdx.x;
  int t = tok[n];
  int pos = n & 511;   // L = 512
#pragma unroll
  for (int j = 0; j < 2; ++j) {
    int d = threadIdx.x + j * 256;
    float div = __expf(-(float)(d & ~1) * 0.017988946f);  // ln(10000)/512
    float pe = (d & 1) ? __cosf((float)pos * div) : __sinf((float)pos * div);
    out[(long long)n * 512 + d] = emb[(long long)t * 512 + d] * 22.62741699796952f + pe;
  }
}

// ---------------------------------------------------------------------------
// x_out = LayerNorm(xin + res) * g + b.  One wave per row (D = 512).
// ---------------------------------------------------------------------------
__global__ void ln_res(const float* __restrict__ xin, const float* __restrict__ res,
                       const float* __restrict__ g, const float* __restrict__ b,
                       float* __restrict__ out)
{
  int row  = blockIdx.x * 8 + (threadIdx.x >> 5);
  int lane = threadIdx.x & 31;
  const float* px = xin + (long long)row * 512;
  const float* pr = res + (long long)row * 512;
  float v[16];
  float s = 0.f;
#pragma unroll
  for (int i = 0; i < 16; ++i) {
    int c = i * 32 + lane;
    v[i] = px[c] + pr[c];
    s += v[i];
  }
#pragma unroll
  for (int off = 16; off > 0; off >>= 1) s += __shfl_xor(s, off, 32);
  float mean = s * (1.f / 512.f);
  float q = 0.f;
#pragma unroll
  for (int i = 0; i < 16; ++i) { float d = v[i] - mean; q += d * d; }
#pragma unroll
  for (int off = 16; off > 0; off >>= 1) q += __shfl_xor(q, off, 32);
  float inv = rsqrtf(q * (1.f / 512.f) + 1e-5f);
#pragma unroll
  for (int i = 0; i < 16; ++i) {
    int c = i * 32 + lane;
    out[(long long)row * 512 + c] = g[c] * (v[i] - mean) * inv + b[c];
  }
}

// ---------------------------------------------------------------------------
// Masked softmax over 512-wide score rows, in place. One wave per row.
// mode 0: src-key mask (tok!=0), mode 1: causal. rows = (b*H + h)*L + qi.
// ---------------------------------------------------------------------------
__global__ void softmax_mask(float* __restrict__ s, const int* __restrict__ srcTok, int mode)
{
  int row  = blockIdx.x * 8 + (threadIdx.x >> 5);
  int lane = threadIdx.x & 31;
  int qi = row & 511;
  int b  = row >> 12;          // row / (H*L) = row / 4096
  float* p = s + (long long)row * 512;
  float r[16];
  float mx = -3.4e38f;
#pragma unroll
  for (int i = 0; i < 16; ++i) {
    int col = i * 32 + lane;
    float v = p[col];
    bool ok = (mode == 0) ? (srcTok[b * 512 + col] != 0) : (col <= qi);
    v = ok ? v : -1e9f;
    r[i] = v;
    mx = fmaxf(mx, v);
  }
#pragma unroll
  for (int off = 16; off > 0; off >>= 1) mx = fmaxf(mx, __shfl_xor(mx, off, 32));
  float sum = 0.f;
#pragma unroll
  for (int i = 0; i < 16; ++i) { r[i] = __expf(r[i] - mx); sum += r[i]; }
#pragma unroll
  for (int off = 16; off > 0; off >>= 1) sum += __shfl_xor(sum, off, 32);
  float inv = 1.f / sum;
#pragma unroll
  for (int i = 0; i < 16; ++i) p[i * 32 + lane] = r[i] * inv;
}

// ---------------------------------------------------------------------------
// Router math for one token (one wave): gates softmax over E=8, top-2.
// ---------------------------------------------------------------------------
__device__ __forceinline__ void router_token(const float* __restrict__ xrow,
                                             const float* __restrict__ wr,
                                             const float* __restrict__ br,
                                             int lane, float gates[8],
                                             int* b0o, int* b1o, float* w0o, float* w1o)
{
  float a[8];
#pragma unroll
  for (int e = 0; e < 8; ++e) a[e] = 0.f;
  for (int c = lane; c < 512; c += 32) {
    float xv = xrow[c];
    const float* w = wr + c * 8;
#pragma unroll
    for (int e = 0; e < 8; ++e) a[e] += xv * w[e];
  }
#pragma unroll
  for (int e = 0; e < 8; ++e)
#pragma unroll
    for (int off = 16; off > 0; off >>= 1) a[e] += __shfl_xor(a[e], off, 32);
  float mx = -3.4e38f;
#pragma unroll
  for (int e = 0; e < 8; ++e) { a[e] += br[e]; mx = fmaxf(mx, a[e]); }
  float s = 0.f;
#pragma unroll
  for (int e = 0; e < 8; ++e) { a[e] = __expf(a[e] - mx); s += a[e]; }
  float inv = 1.f / s;
#pragma unroll
  for (int e = 0; e < 8; ++e) gates[e] = a[e] * inv;
  int b0 = 0;
#pragma unroll
  for (int e = 1; e < 8; ++e) if (gates[e] > gates[b0]) b0 = e;
  int b1 = (b0 == 0) ? 1 : 0;
#pragma unroll
  for (int e = 0; e < 8; ++e) if (e != b0 && gates[e] > gates[b1]) b1 = e;
  float t = gates[b0] + gates[b1];
  *b0o = b0; *b1o = b1;
  *w0o = gates[b0] / t; *w1o = gates[b1] / t;
}

// One wave per token: write renormalized top-2 combine row comb[n][8].
__global__ void router_comb(const float* __restrict__ x, const float* __restrict__ wr,
                            const float* __restrict__ br, float* __restrict__ comb)
{
  int n    = blockIdx.x * 8 + (threadIdx.x >> 5);
  int lane = threadIdx.x & 31;
  float g[8]; int b0, b1; float w0, w1;
  router_token(x + (long long)n * 512, wr, br, lane, g, &b0, &b1, &w0, &w1);
  if (lane < 8)
    comb[n * 8 + lane] = (lane == b0) ? w0 : ((lane == b1) ? w1 : 0.f);
}

// Deterministic single-block load-balance loss: aux += AUX_W * E * sum f_e * P_e.
__global__ void lb_kernel(const float* __restrict__ x, const float* __restrict__ wr,
                          const float* __restrict__ br, float* __restrict__ aux)
{
  __shared__ float sg[32][8];
  __shared__ float sc[32][8];
  int wave = threadIdx.x >> 5;
  int lane = threadIdx.x & 31;
  float lg[8], lc[8];
#pragma unroll
  for (int e = 0; e < 8; ++e) { lg[e] = 0.f; lc[e] = 0.f; }
  for (int n = wave; n < 2048; n += 32) {
    float g[8]; int b0, b1; float w0, w1;
    router_token(x + (long long)n * 512, wr, br, lane, g, &b0, &b1, &w0, &w1);
    if (lane == 0) {
#pragma unroll
      for (int e = 0; e < 8; ++e) lg[e] += g[e];
      lc[b0] += 1.f;
      lc[b1] += 1.f;
    }
  }
  if (lane == 0)
#pragma unroll
    for (int e = 0; e < 8; ++e) { sg[wave][e] = lg[e]; sc[wave][e] = lc[e]; }
  __syncthreads();
  if (threadIdx.x == 0) {
    float contrib = 0.f;
    for (int e = 0; e < 8; ++e) {
      float G = 0.f, C = 0.f;
      for (int w = 0; w < 32; ++w) { G += sg[w][e]; C += sc[w][e]; }
      float f = C / (2048.f * 2.f);
      float P = G / 2048.f;
      contrib += f * P;
    }
    *aux += 8.f * contrib * 0.01f;   // E * sum(f*P) * AUX_W
  }
}

// acc = (first ? 0 : acc) + comb[n][e] * y   over N*D elements.
__global__ void moe_combine(const float* __restrict__ y, const float* __restrict__ comb,
                            float* __restrict__ acc, int e, int first)
{
  long long i = (long long)blockIdx.x * 256 + threadIdx.x;
  int n = (int)(i >> 9);
  float v = comb[n * 8 + e] * y[i];
  acc[i] = first ? v : acc[i] + v;
}

__global__ void zero1(float* p) { *p = 0.f; }

// ---------------------------------------------------------------------------
extern "C" void kernel_launch(void* const* d_in, const int* in_sizes, int n_in,
                              void* d_out, int out_size, void* d_ws, size_t ws_size,
                              hipStream_t stream)
{
  (void)in_sizes; (void)n_in; (void)ws_size;

  const int* src = (const int*)d_in[0];
  const int* tgt = (const int*)d_in[1];
  auto f = [&](int i) { return (const float*)d_in[i]; };
  const float* enc_emb = f(2);
  const float* dec_emb = f(3);

  struct Attn { const float *wq,*bq,*wk,*bk,*wv,*bv,*wo,*bo; };
  struct Moe  { const float *wr,*br,*w1,*b1,*w2,*b2; };
  struct EncL { Attn at; Moe mo; const float *g1,*b1n,*g2,*b2n; };
  struct DecL { Attn at, cr; Moe mo; const float *g1,*b1n,*g2,*b2n,*g3,*b3n; };
  EncL EL[2]; DecL DL[2];
  int ix = 4;
  auto rdA = [&](Attn& a) { a.wq=f(ix++); a.bq=f(ix++); a.wk=f(ix++); a.bk=f(ix++);
                            a.wv=f(ix++); a.bv=f(ix++); a.wo=f(ix++); a.bo=f(ix++); };
  auto rdM = [&](Moe& m)  { m.wr=f(ix++); m.br=f(ix++); m.w1=f(ix++); m.b1=f(ix++);
                            m.w2=f(ix++); m.b2=f(ix++); };
  for (int l = 0; l < 2; ++l) {
    rdA(EL[l].at); rdM(EL[l].mo);
    EL[l].g1=f(ix++); EL[l].b1n=f(ix++); EL[l].g2=f(ix++); EL[l].b2n=f(ix++);
  }
  for (int l = 0; l < 2; ++l) {
    rdA(DL[l].at); rdA(DL[l].cr); rdM(DL[l].mo);
    DL[l].g1=f(ix++); DL[l].b1n=f(ix++); DL[l].g2=f(ix++);
    DL[l].b2n=f(ix++); DL[l].g3=f(ix++); DL[l].b3n=f(ix++);
  }
  const float* fc_w = f(ix++);
  const float* fc_b = f(ix++);

  // workspace layout (floats)
  const long long ND = 2048LL * 512;
  float* W    = (float*)d_ws;
  float* xA   = W + 0 * ND;
  float* xB   = W + 1 * ND;
  float* qb   = W + 2 * ND;
  float* kb   = W + 3 * ND;
  float* vb   = W + 4 * ND;
  float* ob   = W + 5 * ND;
  float* pb   = W + 6 * ND;   // attn projection output
  float* yb   = W + 7 * ND;   // per-expert MoE output
  float* accb = W + 8 * ND;   // combined MoE output
  float* encb = W + 9 * ND;   // saved encoder output
  float* comb = W + 10 * ND;                 // [2048, 8]
  float* sc   = comb + 2048 * 8;             // scores [32, 512, 512]
  float* hb   = sc + 32LL * 512 * 512;       // [2048, 2048]

  float* logits = (float*)d_out;
  float* auxp   = logits + (long long)out_size - 1;

  auto gemm = [&](const float* A, const float* B, const float* bias, float* C,
                  int M, int N, int Kd, int lda, int ldb, int ldc,
                  float alpha, int tB, int relu, int batch, int binner,
                  long long sAo, long long sAi, long long sBo, long long sBi,
                  long long sCo, long long sCi) {
    dim3 g((N + TILE_N - 1) / TILE_N, (M + TILE_M - 1) / TILE_M, batch);
    gemm_wmma_f16<<<g, 256, 0, stream>>>(A, B, bias, C, M, N, Kd, lda, ldb, ldc,
                                         alpha, tB, relu, binner,
                                         sAo, sAi, sBo, sBi, sCo, sCi);
  };

  // Full multi-head attention: out = softmax(mask(QK^T/8)) V Wo + bo
  auto attention = [&](const float* xq, const float* xkv, const Attn& a,
                       int maskmode, float* outbuf) {
    gemm(xq,  a.wq, a.bq, qb, 2048, 512, 512, 512, 512, 512, 1.f, 0, 0, 1, 1, 0,0,0,0,0,0);
    gemm(xkv, a.wk, a.bk, kb, 2048, 512, 512, 512, 512, 512, 1.f, 0, 0, 1, 1, 0,0,0,0,0,0);
    gemm(xkv, a.wv, a.bv, vb, 2048, 512, 512, 512, 512, 512, 1.f, 0, 0, 1, 1, 0,0,0,0,0,0);
    // scores[b,h] = Q_bh (512x64) @ K_bh^T : B given as B^T = K rows
    gemm(qb, kb, nullptr, sc, 512, 512, 64, 512, 512, 512, 0.125f, 1, 0, 32, 8,
         262144, 64, 262144, 64, 2097152, 262144);
    softmax_mask<<<2048, 256, 0, stream>>>(sc, src, maskmode);
    // o[b,h] = attn (512x512) @ V_bh (512x64)
    gemm(sc, vb, nullptr, ob, 512, 64, 512, 512, 512, 512, 1.f, 0, 0, 32, 8,
         2097152, 262144, 262144, 64, 262144, 64);
    gemm(ob, a.wo, a.bo, outbuf, 2048, 512, 512, 512, 512, 512, 1.f, 0, 0, 1, 1, 0,0,0,0,0,0);
  };

  // Dense-expert MoE (mathematically equal to reference's dense compute + sparse combine)
  auto moe = [&](const float* xin, const Moe& m) {
    router_comb<<<256, 256, 0, stream>>>(xin, m.wr, m.br, comb);
    for (int e = 0; e < 8; ++e) {
      gemm(xin, m.w1 + (long long)e * 512 * 2048, m.b1 + e * 2048, hb,
           2048, 2048, 512, 512, 2048, 2048, 1.f, 0, 1, 1, 1, 0,0,0,0,0,0);
      gemm(hb, m.w2 + (long long)e * 2048 * 512, m.b2 + e * 512, yb,
           2048, 512, 2048, 2048, 512, 512, 1.f, 0, 0, 1, 1, 0,0,0,0,0,0);
      moe_combine<<<4096, 256, 0, stream>>>(yb, comb, accb, e, e == 0 ? 1 : 0);
    }
  };

  zero1<<<1, 1, 0, stream>>>(auxp);

  // ---------------- encoder ----------------
  float* x  = xA;
  float* xo = xB;
  embed_pe<<<2048, 256, 0, stream>>>(src, enc_emb, x);
  for (int l = 0; l < 2; ++l) {
    attention(x, x, EL[l].at, /*src mask*/0, pb);
    ln_res<<<256, 256, 0, stream>>>(pb, x, EL[l].g1, EL[l].b1n, xo);
    { float* t = x; x = xo; xo = t; }
    moe(x, EL[l].mo);
    ln_res<<<256, 256, 0, stream>>>(accb, x, EL[l].g2, EL[l].b2n, xo);
    { float* t = x; x = xo; xo = t; }
    lb_kernel<<<1, 1024, 0, stream>>>(x, EL[l].mo.wr, EL[l].mo.br, auxp);
  }
  hipMemcpyAsync(encb, x, ND * sizeof(float), hipMemcpyDeviceToDevice, stream);

  // ---------------- decoder ----------------
  embed_pe<<<2048, 256, 0, stream>>>(tgt, dec_emb, x);
  for (int l = 0; l < 2; ++l) {
    attention(x, x, DL[l].at, /*causal*/1, pb);
    ln_res<<<256, 256, 0, stream>>>(pb, x, DL[l].g1, DL[l].b1n, xo);
    { float* t = x; x = xo; xo = t; }
    attention(x, encb, DL[l].cr, /*src mask*/0, pb);
    ln_res<<<256, 256, 0, stream>>>(pb, x, DL[l].g2, DL[l].b2n, xo);
    { float* t = x; x = xo; xo = t; }
    moe(x, DL[l].mo);
    ln_res<<<256, 256, 0, stream>>>(accb, x, DL[l].g3, DL[l].b3n, xo);
    { float* t = x; x = xo; xo = t; }
    lb_kernel<<<1, 1024, 0, stream>>>(x, DL[l].mo.wr, DL[l].mo.br, auxp);
  }

  // ---------------- final vocab projection ----------------
  gemm(x, fc_w, fc_b, logits, 2048, 32000, 512, 512, 32000, 32000,
       1.f, 0, 0, 1, 1, 0,0,0,0,0,0);
}
